// RAILTEncoder_27513560498693
// MI455X (gfx1250) — compile-verified, matrix operand order checked by reading, not run on hardware
//
#include <hip/hip_runtime.h>
#include <hip/hip_bf16.h>

// ---------------- model constants ----------------
#define D_MODEL   1024
#define N_HEADS   16
#define DH        64
#define CHUNK_LEN 50
#define MEM_LEN   16
#define T_TOT     66          // MEM_LEN + CHUNK_LEN
#define B_SZ      64
#define L_SZ      2000
#define N_CHUNKS  40          // L / CHUNK_LEN
#define ROWS      (B_SZ * T_TOT)   // 4224
#define LN_EPS    1e-5f

typedef __bf16 bf16_t;
typedef __attribute__((ext_vector_type(16))) __bf16 v16bf;
typedef __attribute__((ext_vector_type(8)))  float  v8f;
typedef __attribute__((ext_vector_type(4)))  unsigned v4u;
typedef __attribute__((ext_vector_type(8)))  unsigned v8u;

// ---------------- workspace layout (bytes) ----------------
static constexpr size_t OFF_WQKV  = 0;                                        // bf16 3072x1024
static constexpr size_t OFF_WOUT  = OFF_WQKV  + (size_t)3072*1024*2;          // bf16 1024x1024
static constexpr size_t OFF_W1    = OFF_WOUT  + (size_t)1024*1024*2;          // bf16 4096x1024
static constexpr size_t OFF_W2    = OFF_W1    + (size_t)4096*1024*2;          // bf16 1024x4096
static constexpr size_t OFF_WGATE = OFF_W2    + (size_t)1024*4096*2;          // bf16 1024x1024
static constexpr size_t OFF_Z     = OFF_WGATE + (size_t)1024*1024*2;          // f32  ROWSx1024
static constexpr size_t OFF_LNA   = OFF_Z     + (size_t)ROWS*1024*4;          // bf16 ROWSx1024
static constexpr size_t OFF_QKV   = OFF_LNA   + (size_t)ROWS*1024*2;          // f32  ROWSx3072
static constexpr size_t OFF_O     = OFF_QKV   + (size_t)ROWS*3072*4;          // bf16 ROWSx1024
static constexpr size_t OFF_H     = OFF_O     + (size_t)ROWS*1024*2;          // bf16 ROWSx1024
static constexpr size_t OFF_FFN1  = OFF_H     + (size_t)ROWS*1024*2;          // bf16 ROWSx4096
static constexpr size_t OFF_MCB   = OFF_FFN1  + (size_t)ROWS*4096*2;          // bf16 (B*16)x1024
static constexpr size_t OFF_MEMA  = OFF_MCB   + (size_t)1024*1024*2;          // f32  (B*16)x1024
static constexpr size_t OFF_MEMB  = OFF_MEMA  + (size_t)1024*1024*4;          // f32  (B*16)x1024

// ---------------- CDNA5 async / TDM helpers ----------------
__device__ __forceinline__ void async_load_b128(unsigned lds_off, const bf16_t* g) {
  // GLOBAL_LOAD_ASYNC_TO_LDS_B128: VDST = LDS byte offset, VADDR = 64-bit global addr
  asm volatile("global_load_async_to_lds_b128 %0, %1, off"
               :: "v"(lds_off), "v"((unsigned long long)(uintptr_t)g)
               : "memory");
}
__device__ __forceinline__ void wait_async0() {
  asm volatile("s_wait_asynccnt 0x0" ::: "memory");
}

// TDM: DMA one [rows x 32] bf16 tile (row stride = K elements) into LDS.
// D# group0: count=1 | lds_addr | global_addr(57b) | type=2
// D# group1: data_size=2B, tensor_dim0=K, tensor_dim1=rows, tile_dim0=32,
//            tile_dim1=rows, tensor_dim0_stride=K
__device__ __forceinline__ void tdm_load_tile_b(unsigned lds_off, const bf16_t* g,
                                                int K, int rows) {
  const unsigned long long ga = (unsigned long long)(uintptr_t)g;
  v4u g0;
  g0.x = 1u;                                                  // count=1, user desc
  g0.y = lds_off;                                             // lds_addr
  g0.z = (unsigned)ga;                                        // global_addr[31:0]
  g0.w = (unsigned)((ga >> 32) & 0x01FFFFFFu) | (2u << 30);   // addr[56:32] | type=2
  v8u g1;
  g1.s0 = 0x00010000u;                                        // data_size=1 (2 bytes)
  g1.s1 = ((unsigned)K & 0xFFFFu) << 16;                      // tensor_dim0[15:0]
  g1.s2 = ((unsigned)K >> 16) | (((unsigned)rows & 0xFFFFu) << 16); // dim0 hi | dim1 lo
  g1.s3 = (32u << 16);                                        // dim1 hi=0 | tile_dim0=32
  g1.s4 = (unsigned)rows;                                     // tile_dim1=rows, tile_dim2=0
  g1.s5 = (unsigned)K;                                        // tensor_dim0_stride[31:0]
  g1.s6 = 0u;                                                 // stride hi | dim1_stride lo
  g1.s7 = 0u;
  asm volatile("tensor_load_to_lds %0, %1" :: "s"(g0), "s"(g1) : "memory");
}

// ---------------- helpers ----------------
__global__ __launch_bounds__(256)
void cvt_f32_to_bf16(const float* __restrict__ in, bf16_t* __restrict__ out, int n) {
  int i = blockIdx.x * blockDim.x + threadIdx.x;
  int stride = gridDim.x * blockDim.x;
  for (; i < n; i += stride) out[i] = (bf16_t)in[i];
}

__global__ __launch_bounds__(256)
void mem_bcast(const float* __restrict__ mem_init, float* __restrict__ mem, int n) {
  int i = blockIdx.x * blockDim.x + threadIdx.x;
  int stride = gridDim.x * blockDim.x;
  const int per = MEM_LEN * D_MODEL;
  for (; i < n; i += stride) mem[i] = mem_init[i % per];
}

__global__ __launch_bounds__(256)
void copy_f32(const float* __restrict__ in, float* __restrict__ out, int n) {
  int i = blockIdx.x * blockDim.x + threadIdx.x;
  int stride = gridDim.x * blockDim.x;
  for (; i < n; i += stride) out[i] = in[i];
}

// ---------------- build z (gather + pos) and attention-LN ----------------
__global__ __launch_bounds__(256)
void build_z_ln(const int* __restrict__ in_prob, const int* __restrict__ in_skill,
                const float* __restrict__ emb_p, const float* __restrict__ emb_s,
                const float* __restrict__ mem, const float* __restrict__ pos,
                const float* __restrict__ gamma, const float* __restrict__ beta,
                float* __restrict__ z, bf16_t* __restrict__ ln_out, int chunk) {
  __shared__ float s_sum[256], s_sq[256];
  const int row = blockIdx.x;           // 0..ROWS-1
  const int b = row / T_TOT, t = row % T_TOT;
  const int tid = threadIdx.x;
  const int c0 = tid * 4;
  float v[4];
  if (t < MEM_LEN) {
    const float* m = mem + ((size_t)(b * MEM_LEN + t)) * D_MODEL + c0;
    #pragma unroll
    for (int i = 0; i < 4; ++i) v[i] = m[i];
  } else {
    const int li = chunk * CHUNK_LEN + (t - MEM_LEN);
    const int p = in_prob[(size_t)b * L_SZ + li];
    const int s = in_skill[(size_t)b * L_SZ + li];
    const float* ep = emb_p + (size_t)p * D_MODEL + c0;
    const float* es = emb_s + (size_t)s * D_MODEL + c0;
    #pragma unroll
    for (int i = 0; i < 4; ++i) v[i] = ep[i] + es[i];
  }
  const float* pr = pos + (size_t)t * D_MODEL + c0;
  #pragma unroll
  for (int i = 0; i < 4; ++i) v[i] += pr[i];
  float* zr = z + (size_t)row * D_MODEL + c0;
  #pragma unroll
  for (int i = 0; i < 4; ++i) zr[i] = v[i];

  float lsum = 0.f, lsq = 0.f;
  #pragma unroll
  for (int i = 0; i < 4; ++i) { lsum += v[i]; lsq += v[i] * v[i]; }
  s_sum[tid] = lsum; s_sq[tid] = lsq;
  __syncthreads();
  for (int off = 128; off > 0; off >>= 1) {
    if (tid < off) { s_sum[tid] += s_sum[tid + off]; s_sq[tid] += s_sq[tid + off]; }
    __syncthreads();
  }
  const float mu = s_sum[0] * (1.f / D_MODEL);
  const float var = s_sq[0] * (1.f / D_MODEL) - mu * mu;
  const float rstd = rsqrtf(var + LN_EPS);
  bf16_t* lo = ln_out + (size_t)row * D_MODEL + c0;
  #pragma unroll
  for (int i = 0; i < 4; ++i)
    lo[i] = (bf16_t)((v[i] - mu) * rstd * gamma[c0 + i] + beta[c0 + i]);
}

// ---------------- plain LayerNorm (f32 in -> bf16 out) ----------------
__global__ __launch_bounds__(256)
void ln_kernel(const float* __restrict__ x, const float* __restrict__ gamma,
               const float* __restrict__ beta, bf16_t* __restrict__ out) {
  __shared__ float s_sum[256], s_sq[256];
  const int row = blockIdx.x;
  const int tid = threadIdx.x;
  const int c0 = tid * 4;
  const float* xr = x + (size_t)row * D_MODEL + c0;
  float v[4];
  #pragma unroll
  for (int i = 0; i < 4; ++i) v[i] = xr[i];
  float lsum = 0.f, lsq = 0.f;
  #pragma unroll
  for (int i = 0; i < 4; ++i) { lsum += v[i]; lsq += v[i] * v[i]; }
  s_sum[tid] = lsum; s_sq[tid] = lsq;
  __syncthreads();
  for (int off = 128; off > 0; off >>= 1) {
    if (tid < off) { s_sum[tid] += s_sum[tid + off]; s_sq[tid] += s_sq[tid + off]; }
    __syncthreads();
  }
  const float mu = s_sum[0] * (1.f / D_MODEL);
  const float var = s_sq[0] * (1.f / D_MODEL) - mu * mu;
  const float rstd = rsqrtf(var + LN_EPS);
  bf16_t* o = out + (size_t)row * D_MODEL + c0;
  #pragma unroll
  for (int i = 0; i < 4; ++i)
    o[i] = (bf16_t)((v[i] - mu) * rstd * gamma[c0 + i] + beta[c0 + i]);
}

// ---------------- WMMA bf16 GEMM:  C[M,N] = A[M,K] * W[N,K]^T (+ epilogue) ----------------
// A tile: per-lane GLOBAL_LOAD_ASYNC_TO_LDS_B128 (ASYNCcnt)
// B tile: one TDM tensor_load_to_lds per workgroup (TENSORcnt, issued by wave 0)
// EPI 0: Cf = acc + bias                       (f32 out)
// EPI 1: Cb = relu(acc + bias)                 (bf16 out)
// EPI 2: Cf = acc + bias + resid               (f32 out, residual add)
// EPI 3: gated memory update (sigmoid gate)    (f32 out)
static constexpr int BM = 128, BN = 128, BK = 32;

template <int EPI>
__global__ __launch_bounds__(256)
void gemm_bf16_wmma(const bf16_t* __restrict__ A, const bf16_t* __restrict__ W,
                    const float* __restrict__ bias,
                    float* __restrict__ Cf, bf16_t* __restrict__ Cb,
                    const float* __restrict__ resid, const float* __restrict__ memprev,
                    int M, int N, int K, int first_chunk) {
  __shared__ __align__(64) bf16_t sA[2][BM * BK];   // 8 KB x2
  __shared__ __align__(64) bf16_t sB[2][BN * BK];   // 8 KB x2

  const int tid = threadIdx.x;
  const int m0 = blockIdx.x * BM;
  const int n0 = blockIdx.y * BN;
  const int wave = tid >> 5, lane = tid & 31;
  const int wm = wave >> 1, wn = wave & 1;       // 4 x 2 wave grid -> 32x64 per wave
  const int l16 = lane & 15, lh = lane >> 4;     // fragment row/col and K-half

  v8f acc[2][4];
  #pragma unroll
  for (int mi = 0; mi < 2; ++mi)
    #pragma unroll
    for (int ni = 0; ni < 4; ++ni)
      acc[mi][ni] = (v8f){0.f, 0.f, 0.f, 0.f, 0.f, 0.f, 0.f, 0.f};

  // cooperative async fill of A tile: 256 threads x 32B
  const int tr = tid >> 1;               // 0..127 row
  const int tc = (tid & 1) << 4;         // 0 or 16 (bf16 elements)
  const int nk = K / BK;

  auto issue_tile_a = [&](int buf, int kb) {
    const bf16_t* ag = A + (size_t)(m0 + tr) * K + kb + tc;
    async_load_b128((unsigned)(uintptr_t)&sA[buf][tr * BK + tc],     ag);
    async_load_b128((unsigned)(uintptr_t)&sA[buf][tr * BK + tc + 8], ag + 8);
  };
  auto issue_tile_b = [&](int buf, int kb) {   // wave 0 only: TDM DMA of the weight tile
    tdm_load_tile_b((unsigned)(uintptr_t)&sB[buf][0],
                    W + (size_t)n0 * K + kb, K, BN);
  };

  if (wave == 0) issue_tile_b(0, 0);
  issue_tile_a(0, 0);

  for (int kt = 0; kt < nk; ++kt) {
    wait_async0();                                 // own A-tile async stores landed
    if (wave == 0) __builtin_amdgcn_s_wait_tensorcnt(0);  // B-tile DMA landed
    __syncthreads();                               // publish to all waves
    const int cur = kt & 1;
    if (kt + 1 < nk) {
      const int kb = (kt + 1) * BK;
      if (wave == 0) issue_tile_b(cur ^ 1, kb);
      issue_tile_a(cur ^ 1, kb);
      if (kt + 2 < nk) {  // L2 prefetch A two tiles ahead
        __builtin_prefetch(A + (size_t)(m0 + tr) * K + (kt + 2) * BK + tc, 0, 1);
      }
    }

    v16bf afrag[2], bfrag[4];
    #pragma unroll
    for (int mi = 0; mi < 2; ++mi)
      afrag[mi] = *(const v16bf*)(&sA[cur][(wm * 32 + mi * 16 + l16) * BK + lh * 16]);
    #pragma unroll
    for (int ni = 0; ni < 4; ++ni)
      bfrag[ni] = *(const v16bf*)(&sB[cur][(wn * 64 + ni * 16 + l16) * BK + lh * 16]);

    #pragma unroll
    for (int mi = 0; mi < 2; ++mi)
      #pragma unroll
      for (int ni = 0; ni < 4; ++ni)
        acc[mi][ni] = __builtin_amdgcn_wmma_f32_16x16x32_bf16(
            false, afrag[mi], false, bfrag[ni], (short)0, acc[mi][ni], false, false);
  }

  // epilogue: lane l holds column N = l16; VGPR r holds row M = 8*lh + r
  #pragma unroll
  for (int mi = 0; mi < 2; ++mi) {
    #pragma unroll
    for (int ni = 0; ni < 4; ++ni) {
      const int gn = n0 + wn * 64 + ni * 16 + l16;
      const float bv = bias[gn];
      #pragma unroll
      for (int r = 0; r < 8; ++r) {
        const int gm = m0 + wm * 32 + mi * 16 + lh * 8 + r;
        const size_t off = (size_t)gm * N + gn;
        float v = acc[mi][ni][r] + bv;
        if (EPI == 0) {
          Cf[off] = v;
        } else if (EPI == 1) {
          Cb[off] = (bf16_t)(v > 0.f ? v : 0.f);
        } else if (EPI == 2) {
          Cf[off] = v + resid[off];
        } else {
          // gate: m = b*MEM_LEN + t ; candidate lives in z at row b*T_TOT + t
          const int bb = gm >> 4, tt = gm & 15;
          const size_t zoff = ((size_t)(bb * T_TOT + tt)) * N + gn;
          const float cand = resid[zoff];
          const float prev = memprev[off];
          const float g = 1.f / (1.f + __expf(-v));
          Cf[off] = first_chunk ? cand : (g * cand + (1.f - g) * prev);
        }
      }
    }
  }
}

// ---------------- attention (per batch*head), streaming softmax ----------------
__global__ __launch_bounds__(128)
void attn_kernel(const float* __restrict__ qkv, bf16_t* __restrict__ o) {
  __shared__ float sk[T_TOT * DH];
  __shared__ float sv[T_TOT * DH];
  const int bh = blockIdx.x;
  const int b = bh >> 4, h = bh & 15;
  const int tid = threadIdx.x;
  const size_t rs = 3 * D_MODEL;
  const float* basek = qkv + (size_t)b * T_TOT * rs + D_MODEL + h * DH;
  const float* basev = qkv + (size_t)b * T_TOT * rs + 2 * D_MODEL + h * DH;
  for (int i = tid; i < T_TOT * DH; i += 128) {
    const int t = i >> 6, c = i & 63;
    sk[i] = basek[(size_t)t * rs + c];
    sv[i] = basev[(size_t)t * rs + c];
  }
  __syncthreads();
  if (tid < T_TOT) {
    const int t = tid;
    const float* qrow = qkv + ((size_t)(b * T_TOT + t)) * rs + h * DH;
    float4 q[16];
    #pragma unroll
    for (int i = 0; i < 16; ++i) q[i] = ((const float4*)qrow)[i];

    // pass 1: online max & normalizer
    float mmax = -1e30f, ssum = 0.f;
    for (int j = 0; j < T_TOT; ++j) {
      const float4* kr = (const float4*)(sk + j * DH);
      float d = 0.f;
      #pragma unroll
      for (int i = 0; i < 16; ++i) {
        const float4 kv = kr[i];
        d += q[i].x * kv.x + q[i].y * kv.y + q[i].z * kv.z + q[i].w * kv.w;
      }
      d *= 0.125f;
      if (t >= MEM_LEN && j > t) d = -1e9f;
      const float nm = fmaxf(mmax, d);
      ssum = ssum * __expf(mmax - nm) + __expf(d - nm);
      mmax = nm;
    }
    const float inv = 1.f / ssum;

    // pass 2: weighted sum of V
    float4 oacc[16];
    #pragma unroll
    for (int i = 0; i < 16; ++i) oacc[i] = (float4){0.f, 0.f, 0.f, 0.f};
    for (int j = 0; j < T_TOT; ++j) {
      const float4* kr = (const float4*)(sk + j * DH);
      float d = 0.f;
      #pragma unroll
      for (int i = 0; i < 16; ++i) {
        const float4 kv = kr[i];
        d += q[i].x * kv.x + q[i].y * kv.y + q[i].z * kv.z + q[i].w * kv.w;
      }
      d *= 0.125f;
      if (t >= MEM_LEN && j > t) d = -1e9f;
      const float a = __expf(d - mmax) * inv;
      const float4* vr = (const float4*)(sv + j * DH);
      #pragma unroll
      for (int i = 0; i < 16; ++i) {
        const float4 vv = vr[i];
        oacc[i].x += a * vv.x; oacc[i].y += a * vv.y;
        oacc[i].z += a * vv.z; oacc[i].w += a * vv.w;
      }
    }
    bf16_t* orow = o + ((size_t)(b * T_TOT + t)) * D_MODEL + h * DH;
    #pragma unroll
    for (int i = 0; i < 16; ++i) {
      orow[i * 4 + 0] = (bf16_t)oacc[i].x;
      orow[i * 4 + 1] = (bf16_t)oacc[i].y;
      orow[i * 4 + 2] = (bf16_t)oacc[i].z;
      orow[i * 4 + 3] = (bf16_t)oacc[i].w;
    }
  }
}

// ---------------- extract: write enc_out rows; bf16 mem-candidate ----------------
__global__ __launch_bounds__(256)
void extract_kernel(const float* __restrict__ z, float* __restrict__ enc,
                    bf16_t* __restrict__ memcand_b, int chunk) {
  const int row = blockIdx.x;
  const int b = row / T_TOT, t = row % T_TOT;
  const int c0 = threadIdx.x * 4;
  const float* zr = z + (size_t)row * D_MODEL + c0;
  if (t < MEM_LEN) {
    bf16_t* mb = memcand_b + ((size_t)(b * MEM_LEN + t)) * D_MODEL + c0;
    #pragma unroll
    for (int i = 0; i < 4; ++i) mb[i] = (bf16_t)zr[i];
  } else {
    float* er = enc + (((size_t)b * L_SZ) + (size_t)chunk * CHUNK_LEN + (t - MEM_LEN)) * D_MODEL + c0;
    #pragma unroll
    for (int i = 0; i < 4; ++i) er[i] = zr[i];
  }
}

// ---------------- host-side launcher ----------------
extern "C" void kernel_launch(void* const* d_in, const int* in_sizes, int n_in,
                              void* d_out, int out_size, void* d_ws, size_t ws_size,
                              hipStream_t stream) {
  const int*   in_prob  = (const int*)d_in[0];
  const int*   in_skill = (const int*)d_in[1];
  const float* emb_p    = (const float*)d_in[2];
  const float* emb_s    = (const float*)d_in[3];
  const float* mem_init = (const float*)d_in[4];
  const float* pos      = (const float*)d_in[5];
  const float* w_qkv    = (const float*)d_in[6];
  const float* b_qkv    = (const float*)d_in[7];
  const float* w_out    = (const float*)d_in[8];
  const float* b_out    = (const float*)d_in[9];
  const float* w1       = (const float*)d_in[10];
  const float* b1       = (const float*)d_in[11];
  const float* w2       = (const float*)d_in[12];
  const float* b2       = (const float*)d_in[13];
  const float* g_attn   = (const float*)d_in[14];
  const float* be_attn  = (const float*)d_in[15];
  const float* g_ffn    = (const float*)d_in[16];
  const float* be_ffn   = (const float*)d_in[17];
  const float* w_gate   = (const float*)d_in[18];
  const float* b_gate   = (const float*)d_in[19];

  float* out = (float*)d_out;                 // enc_out [B,L,D] then mem_final [B,16,D]
  char*  ws  = (char*)d_ws;

  bf16_t* wqkv_b  = (bf16_t*)(ws + OFF_WQKV);
  bf16_t* wout_b  = (bf16_t*)(ws + OFF_WOUT);
  bf16_t* w1_b    = (bf16_t*)(ws + OFF_W1);
  bf16_t* w2_b    = (bf16_t*)(ws + OFF_W2);
  bf16_t* wgate_b = (bf16_t*)(ws + OFF_WGATE);
  float*  z       = (float*)(ws + OFF_Z);
  bf16_t* ln_a    = (bf16_t*)(ws + OFF_LNA);
  float*  qkvf    = (float*)(ws + OFF_QKV);
  bf16_t* o_b     = (bf16_t*)(ws + OFF_O);
  bf16_t* h_b     = (bf16_t*)(ws + OFF_H);
  bf16_t* ffn1_b  = (bf16_t*)(ws + OFF_FFN1);
  bf16_t* mcand_b = (bf16_t*)(ws + OFF_MCB);
  float*  memA    = (float*)(ws + OFF_MEMA);
  float*  memB    = (float*)(ws + OFF_MEMB);

  // one-time: bf16 weights + broadcast mem_init
  cvt_f32_to_bf16<<<4096, 256, 0, stream>>>(w_qkv,  wqkv_b,  3072 * 1024);
  cvt_f32_to_bf16<<<4096, 256, 0, stream>>>(w_out,  wout_b,  1024 * 1024);
  cvt_f32_to_bf16<<<4096, 256, 0, stream>>>(w1,     w1_b,    4096 * 1024);
  cvt_f32_to_bf16<<<4096, 256, 0, stream>>>(w2,     w2_b,    1024 * 4096);
  cvt_f32_to_bf16<<<4096, 256, 0, stream>>>(w_gate, wgate_b, 1024 * 1024);
  mem_bcast<<<4096, 256, 0, stream>>>(mem_init, memA, B_SZ * MEM_LEN * D_MODEL);

  float* memCur = memA;
  float* memNext = memB;

  for (int c = 0; c < N_CHUNKS; ++c) {
    build_z_ln<<<ROWS, 256, 0, stream>>>(in_prob, in_skill, emb_p, emb_s,
                                         memCur, pos, g_attn, be_attn, z, ln_a, c);
    // QKV: [4224,3072] = ln_a[4224,1024] x wqkv[3072,1024]^T
    gemm_bf16_wmma<0><<<dim3(ROWS / BM, 3072 / BN), 256, 0, stream>>>(
        ln_a, wqkv_b, b_qkv, qkvf, nullptr, nullptr, nullptr, ROWS, 3072, 1024, 0);
    attn_kernel<<<B_SZ * N_HEADS, 128, 0, stream>>>(qkvf, o_b);
    // out-proj + residual into z
    gemm_bf16_wmma<2><<<dim3(ROWS / BM, 1024 / BN), 256, 0, stream>>>(
        o_b, wout_b, b_out, z, nullptr, z, nullptr, ROWS, 1024, 1024, 0);
    ln_kernel<<<ROWS, 256, 0, stream>>>(z, g_ffn, be_ffn, h_b);
    // FFN1 with fused relu -> bf16
    gemm_bf16_wmma<1><<<dim3(ROWS / BM, 4096 / BN), 256, 0, stream>>>(
        h_b, w1_b, b1, nullptr, ffn1_b, nullptr, nullptr, ROWS, 4096, 1024, 0);
    // FFN2 + residual into z
    gemm_bf16_wmma<2><<<dim3(ROWS / BM, 1024 / BN), 256, 0, stream>>>(
        ffn1_b, w2_b, b2, z, nullptr, z, nullptr, ROWS, 1024, 4096, 0);
    extract_kernel<<<ROWS, 256, 0, stream>>>(z, out, mcand_b, c);
    // gated memory update: logits = mcand @ w_gate^T + b_gate
    gemm_bf16_wmma<3><<<dim3((B_SZ * MEM_LEN) / BM, 1024 / BN), 256, 0, stream>>>(
        mcand_b, wgate_b, b_gate, memNext, nullptr, z, memCur,
        B_SZ * MEM_LEN, 1024, 1024, (c == 0) ? 1 : 0);
    float* tmp = memCur; memCur = memNext; memNext = tmp;
  }

  // mem_final -> tail of d_out
  copy_f32<<<4096, 256, 0, stream>>>(memCur, out + (size_t)B_SZ * L_SZ * D_MODEL,
                                     B_SZ * MEM_LEN * D_MODEL);
}
